// PerceiverAR_18064632447489
// MI455X (gfx1250) — matrix operand order repaired and checked by reading, not compiled, not run on hardware
//
#include <hip/hip_runtime.h>
#include <hip/hip_bf16.h>
#include <stdint.h>

// ---------------------------------------------------------------------------
// PerceiverAR forward for MI455X (gfx1250), bf16 WMMA GEMMs + fp32 everything
// else. All matmuls lower to V_WMMA_F32_16X16X32_BF16.
// ---------------------------------------------------------------------------

typedef __bf16 bf16_t;
typedef __attribute__((ext_vector_type(16))) __bf16 v16bf;
typedef __attribute__((ext_vector_type(8)))  float  v8f;

#define TB 256

// model dims (match reference)
#define NTOK 4096
#define NLAT 1024
#define NPRE 3072
#define DIM  1024
#define NH   16
#define DHD  64
#define FFD  4096
#define VOC  32000
#define NDEPTH 4

// gemm epilogue flags
#define F_BIAS  1
#define F_RES   2
#define F_GELU  4
#define F_BF16  8
#define F_BF16T 16

__device__ __forceinline__ bf16_t f2bf(float x) {
  union { float f; uint32_t u; } c; c.f = x;
  uint32_t r = (c.u + 0x7fffu + ((c.u >> 16) & 1u)) >> 16; // RNE
  union { uint16_t s; bf16_t b; } o; o.s = (uint16_t)r;
  return o.b;
}

struct __attribute__((aligned(16))) U4 { uint32_t x, y, z, w; };

union FragU { v16bf v; U4 q[2]; };

// C = A(MxK, bf16 row-major) * B^T where B stored (N x K) bf16 row-major.
// fp32 accumulate; fused bias / residual / exact-GELU / bf16 (opt transposed).
//
// Fragment layouts (ISA 7.12.2), per lane: m/n = lane&15, half = lane>>4.
// A 16x32: VGPR0-3 K=8*half+0..7, VGPR4-7 K=16+8*half+0..7
//          -> U4 chunks [half] and [2+half] of the 32-element row slice.
// B 32x16: same K order as A: K=8*half+0..7, then K=16+8*half+0..7
//          -> U4 chunks [half] and [2+half].
// All M launched are multiples of 128 so A rows never need a guard. B rows
// (columns of C) are clamped to N-1: out-of-range columns compute garbage
// that the epilogue never stores.
__global__ __launch_bounds__(TB)
void gemm_nt_kernel(const bf16_t* __restrict__ A, int lda,
                    const bf16_t* __restrict__ B, int ldb,
                    void* __restrict__ Cout, int ldc,
                    const float* __restrict__ bias,
                    const float* __restrict__ resid, int ldr,
                    int M, int N, int K, int flags) {
  const int lane = threadIdx.x & 31;
  const int half = lane >> 4;
  const int wid  = threadIdx.x >> 5;
  const int wm = blockIdx.y * 128 + (wid & 3) * 32;  // wave row base
  const int wn = blockIdx.x * 128 + (wid >> 2) * 64; // wave col base

  // hoisted per-fragment row base pointers (U4 = 8 bf16)
  const int ar0 = wm + (lane & 15);
  const int ar1 = ar0 + 16;
  auto bclamp = [&](int n) { return n < N ? n : N - 1; };
  const int bn0 = bclamp(wn +      (lane & 15));
  const int bn1 = bclamp(wn + 16 + (lane & 15));
  const int bn2 = bclamp(wn + 32 + (lane & 15));
  const int bn3 = bclamp(wn + 48 + (lane & 15));
  const U4* a0p = (const U4*)(A + (size_t)ar0 * lda);
  const U4* a1p = (const U4*)(A + (size_t)ar1 * lda);
  const U4* b0p = (const U4*)(B + (size_t)bn0 * ldb);
  const U4* b1p = (const U4*)(B + (size_t)bn1 * ldb);
  const U4* b2p = (const U4*)(B + (size_t)bn2 * ldb);
  const U4* b3p = (const U4*)(B + (size_t)bn3 * ldb);

  v8f zero = {};
  v8f acc[2][4];
#pragma unroll
  for (int i = 0; i < 2; ++i)
#pragma unroll
    for (int j = 0; j < 4; ++j) acc[i][j] = zero;

  for (int ks = 0; ks < K; ks += 32) {
    const int kq = ks >> 3; // U4 index of this K tile
    // speculative prefetch of next K tile (global_prefetch_b8; OOB is dropped)
    __builtin_prefetch((const void*)(a0p + kq + 4), 0, 1);
    __builtin_prefetch((const void*)(b0p + kq + 4), 0, 1);

    FragU a0, a1, b0, b1, b2, b3;
    a0.q[0] = a0p[kq + half];     a0.q[1] = a0p[kq + 2 + half];
    a1.q[0] = a1p[kq + half];     a1.q[1] = a1p[kq + 2 + half];
    b0.q[0] = b0p[kq + half];     b0.q[1] = b0p[kq + 2 + half];
    b1.q[0] = b1p[kq + half];     b1.q[1] = b1p[kq + 2 + half];
    b2.q[0] = b2p[kq + half];     b2.q[1] = b2p[kq + 2 + half];
    b3.q[0] = b3p[kq + half];     b3.q[1] = b3p[kq + 2 + half];

    acc[0][0] = __builtin_amdgcn_wmma_f32_16x16x32_bf16(false, a0.v, false, b0.v, (short)0, acc[0][0], false, false);
    acc[0][1] = __builtin_amdgcn_wmma_f32_16x16x32_bf16(false, a0.v, false, b1.v, (short)0, acc[0][1], false, false);
    acc[0][2] = __builtin_amdgcn_wmma_f32_16x16x32_bf16(false, a0.v, false, b2.v, (short)0, acc[0][2], false, false);
    acc[0][3] = __builtin_amdgcn_wmma_f32_16x16x32_bf16(false, a0.v, false, b3.v, (short)0, acc[0][3], false, false);
    acc[1][0] = __builtin_amdgcn_wmma_f32_16x16x32_bf16(false, a1.v, false, b0.v, (short)0, acc[1][0], false, false);
    acc[1][1] = __builtin_amdgcn_wmma_f32_16x16x32_bf16(false, a1.v, false, b1.v, (short)0, acc[1][1], false, false);
    acc[1][2] = __builtin_amdgcn_wmma_f32_16x16x32_bf16(false, a1.v, false, b2.v, (short)0, acc[1][2], false, false);
    acc[1][3] = __builtin_amdgcn_wmma_f32_16x16x32_bf16(false, a1.v, false, b3.v, (short)0, acc[1][3], false, false);
  }

  // C/D layout: lane&15 = column, lanes>=16 -> row += 8, VGPR v = row offset.
#pragma unroll
  for (int i = 0; i < 2; ++i) {
#pragma unroll
    for (int j = 0; j < 4; ++j) {
      int col = wn + 16 * j + (lane & 15);
      if (col >= N) continue;
      int rbase = wm + 16 * i + (half ? 8 : 0);
      float bv = (flags & F_BIAS) ? bias[col] : 0.0f;
#pragma unroll
      for (int v = 0; v < 8; ++v) {
        int row = rbase + v;
        if (row >= M) continue;
        float c = acc[i][j][v] + bv;
        if (flags & F_RES)  c += resid[(size_t)row * ldr + col];
        if (flags & F_GELU) c = 0.5f * c * (1.0f + erff(c * 0.70710678118654752f));
        if (flags & F_BF16)
          ((bf16_t*)Cout)[(size_t)row * ldc + col] = f2bf(c);
        else if (flags & F_BF16T)
          ((bf16_t*)Cout)[(size_t)col * ldc + row] = f2bf(c); // emits V^T directly
        else
          ((float*)Cout)[(size_t)row * ldc + col] = c;
      }
    }
  }
}

__global__ __launch_bounds__(TB)
void embed_kernel(const int* __restrict__ x, const float* __restrict__ tok,
                  const float* __restrict__ pos, float* __restrict__ e, int D) {
  int t = blockIdx.x;
  int id = x[t];
  const float* ts = tok + (size_t)id * D;
  const float* ps = pos + (size_t)t * D;
  float* es = e + (size_t)t * D;
  for (int d = threadIdx.x; d < D; d += TB) es[d] = ts[d] + ps[d];
}

__global__ __launch_bounds__(TB)
void cvt_bf16_kernel(const float* __restrict__ X, bf16_t* __restrict__ Y, size_t n) {
  size_t i = (size_t)blockIdx.x * TB + threadIdx.x;
  if (i < n) Y[i] = f2bf(X[i]);
}

// W (K x N, f32 row-major) -> WT (N x K, bf16 row-major)
__global__ __launch_bounds__(TB)
void transpose_cvt_kernel(const float* __restrict__ W, bf16_t* __restrict__ WT,
                          int K, int N) {
  size_t idx = (size_t)blockIdx.x * TB + threadIdx.x;
  size_t total = (size_t)K * N;
  if (idx >= total) return;
  int k = (int)(idx / N);
  int n = (int)(idx - (size_t)k * N);
  WT[(size_t)n * K + k] = f2bf(W[idx]);
}

__global__ __launch_bounds__(TB)
void ln_bf16_kernel(const float* __restrict__ X, const float* __restrict__ g,
                    const float* __restrict__ b, bf16_t* __restrict__ Y, int D) {
  __shared__ float s1[TB], s2[TB];
  int row = blockIdx.x;
  const float* x = X + (size_t)row * D;
  float sum = 0.f, sq = 0.f;
  for (int d = threadIdx.x; d < D; d += TB) { float v = x[d]; sum += v; sq += v * v; }
  s1[threadIdx.x] = sum; s2[threadIdx.x] = sq;
  __syncthreads();
  for (int s = TB / 2; s > 0; s >>= 1) {
    if ((int)threadIdx.x < s) { s1[threadIdx.x] += s1[threadIdx.x + s]; s2[threadIdx.x] += s2[threadIdx.x + s]; }
    __syncthreads();
  }
  float mean = s1[0] / D;
  float var  = s2[0] / D - mean * mean;
  float inv  = rsqrtf(var + 1e-5f);
  for (int d = threadIdx.x; d < D; d += TB)
    Y[(size_t)row * D + d] = f2bf((x[d] - mean) * inv * g[d] + b[d]);
}

// row i may attend to j <= i + offset; scale applied pre-softmax (== q*DH^-.5)
__global__ __launch_bounds__(TB)
void softmax_bf16_kernel(const float* __restrict__ S, bf16_t* __restrict__ P,
                         int Nk, int offset, float scale) {
  __shared__ float red[TB];
  int row = blockIdx.x;
  int limit = row + offset;
  const float* s = S + (size_t)row * Nk;
  bf16_t* p = P + (size_t)row * Nk;
  float mx = -3.4e38f;
  for (int j = threadIdx.x; j < Nk; j += TB)
    if (j <= limit) mx = fmaxf(mx, s[j] * scale);
  red[threadIdx.x] = mx; __syncthreads();
  for (int t = TB / 2; t > 0; t >>= 1) {
    if ((int)threadIdx.x < t) red[threadIdx.x] = fmaxf(red[threadIdx.x], red[threadIdx.x + t]);
    __syncthreads();
  }
  float m = red[0]; __syncthreads();
  float sum = 0.f;
  for (int j = threadIdx.x; j < Nk; j += TB)
    if (j <= limit) sum += __expf(s[j] * scale - m);
  red[threadIdx.x] = sum; __syncthreads();
  for (int t = TB / 2; t > 0; t >>= 1) {
    if ((int)threadIdx.x < t) red[threadIdx.x] += red[threadIdx.x + t];
    __syncthreads();
  }
  float inv = 1.0f / red[0];
  for (int j = threadIdx.x; j < Nk; j += TB)
    p[j] = f2bf(j <= limit ? __expf(s[j] * scale - m) * inv : 0.0f);
}

// ---------------------------------------------------------------------------

extern "C" void kernel_launch(void* const* d_in, const int* in_sizes, int n_in,
                              void* d_out, int out_size, void* d_ws, size_t ws_size,
                              hipStream_t stream) {
  (void)in_sizes; (void)n_in; (void)out_size; (void)ws_size;

  const int*   x_ids   = (const int*)  d_in[0];
  const float* tok_emb = (const float*)d_in[1];
  const float* pos_emb = (const float*)d_in[2];
  const float* ca_ln_g = (const float*)d_in[3];
  const float* ca_ln_b = (const float*)d_in[4];
  const float* ca_wq   = (const float*)d_in[5];
  const float* ca_wkv  = (const float*)d_in[6];
  const float* ca_wo   = (const float*)d_in[7];
  const float* ca_bo   = (const float*)d_in[8];
  const float* cf_ln_g = (const float*)d_in[9];
  const float* cf_ln_b = (const float*)d_in[10];
  const float* cf_w1   = (const float*)d_in[11];
  const float* cf_w2   = (const float*)d_in[12];
  const float* la_ln_g = (const float*)d_in[13];
  const float* la_ln_b = (const float*)d_in[14];
  const float* la_wqkv = (const float*)d_in[15];
  const float* la_wo   = (const float*)d_in[16];
  const float* lf_ln_g = (const float*)d_in[17];
  const float* lf_ln_b = (const float*)d_in[18];
  const float* lf_w1   = (const float*)d_in[19];
  const float* lf_w2   = (const float*)d_in[20];
  const float* w_logit = (const float*)d_in[21];

  // ---- workspace layout (~273 MB) ----
  char* wsp = (char*)d_ws;
  size_t off = 0;
  auto alloc = [&](size_t bytes) { size_t o = off; off += (bytes + 255) & ~(size_t)255; return o; };
  size_t o_e    = alloc((size_t)NTOK * DIM * 4);      // embeddings (fp32); rows 3072.. = latent h
  size_t o_pre  = alloc((size_t)NPRE * DIM * 2);      // prefix bf16
  size_t o_ln   = alloc((size_t)NLAT * DIM * 2);      // LN(h) bf16
  size_t o_q    = alloc((size_t)NLAT * DIM * 2);      // Q bf16
  size_t o_k    = alloc((size_t)NTOK * DIM * 2);      // K (rows=keys) bf16
  size_t o_vT   = alloc((size_t)DIM * NTOK * 2);      // V^T (rows=inner dim) bf16
  size_t o_S    = alloc((size_t)NLAT * NTOK * 4);     // per-head scores fp32
  size_t o_P    = alloc((size_t)NLAT * NTOK * 2);     // per-head probs bf16
  size_t o_attn = alloc((size_t)NLAT * DIM * 2);      // attn out bf16
  size_t o_ff1  = alloc((size_t)NLAT * FFD * 2);      // FFN hidden bf16
  size_t o_hbf  = alloc((size_t)NLAT * DIM * 2);      // final h bf16
  size_t o_wqT   = alloc((size_t)DIM * DIM * 2);
  size_t o_wkvT  = alloc((size_t)2 * DIM * DIM * 2);
  size_t o_woT   = alloc((size_t)DIM * DIM * 2);
  size_t o_w1T   = alloc((size_t)FFD * DIM * 2);
  size_t o_w2T   = alloc((size_t)DIM * FFD * 2);
  size_t o_lqkvT = alloc((size_t)NDEPTH * 3 * DIM * DIM * 2);
  size_t o_lwoT  = alloc((size_t)NDEPTH * DIM * DIM * 2);
  size_t o_lw1T  = alloc((size_t)NDEPTH * FFD * DIM * 2);
  size_t o_lw2T  = alloc((size_t)NDEPTH * DIM * FFD * 2);
  size_t o_wlT   = alloc((size_t)VOC * DIM * 2);

  float*  e_f32  = (float*) (wsp + o_e);
  float*  hbuf   = e_f32 + (size_t)NPRE * DIM;        // latent residual stream (in-place)
  bf16_t* pre_bf = (bf16_t*)(wsp + o_pre);
  bf16_t* ln_bf  = (bf16_t*)(wsp + o_ln);
  bf16_t* q_bf   = (bf16_t*)(wsp + o_q);
  bf16_t* k_all  = (bf16_t*)(wsp + o_k);
  bf16_t* vT     = (bf16_t*)(wsp + o_vT);
  float*  Sf     = (float*) (wsp + o_S);
  bf16_t* Pbf    = (bf16_t*)(wsp + o_P);
  bf16_t* attn_bf= (bf16_t*)(wsp + o_attn);
  bf16_t* ff1_bf = (bf16_t*)(wsp + o_ff1);
  bf16_t* h_bf   = (bf16_t*)(wsp + o_hbf);
  bf16_t* wqT    = (bf16_t*)(wsp + o_wqT);
  bf16_t* wkvT   = (bf16_t*)(wsp + o_wkvT);
  bf16_t* woT    = (bf16_t*)(wsp + o_woT);
  bf16_t* w1T    = (bf16_t*)(wsp + o_w1T);
  bf16_t* w2T    = (bf16_t*)(wsp + o_w2T);
  bf16_t* lqkvT  = (bf16_t*)(wsp + o_lqkvT);
  bf16_t* lwoT   = (bf16_t*)(wsp + o_lwoT);
  bf16_t* lw1T   = (bf16_t*)(wsp + o_lw1T);
  bf16_t* lw2T   = (bf16_t*)(wsp + o_lw2T);
  bf16_t* wlT    = (bf16_t*)(wsp + o_wlT);

  auto gemm = [&](const bf16_t* A, int lda, const bf16_t* B, int ldb,
                  void* C, int ldc, const float* bias, const float* resid, int ldr,
                  int M, int Nc, int K, int flags) {
    dim3 grid((unsigned)((Nc + 127) / 128), (unsigned)((M + 127) / 128));
    gemm_nt_kernel<<<grid, TB, 0, stream>>>(A, lda, B, ldb, C, ldc, bias, resid, ldr, M, Nc, K, flags);
  };
  auto tcvt = [&](const float* src, bf16_t* dst, int K, int Ncols) {
    size_t n = (size_t)K * Ncols;
    transpose_cvt_kernel<<<(unsigned)((n + TB - 1) / TB), TB, 0, stream>>>(src, dst, K, Ncols);
  };

  // ---- one-time weight transpose+convert to bf16 (N x K) ----
  tcvt(ca_wq,  wqT,  DIM, DIM);
  tcvt(ca_wkv, wkvT, DIM, 2 * DIM);
  tcvt(ca_wo,  woT,  DIM, DIM);
  tcvt(cf_w1,  w1T,  DIM, FFD);
  tcvt(cf_w2,  w2T,  FFD, DIM);
  for (int l = 0; l < NDEPTH; ++l) {
    tcvt(la_wqkv + (size_t)l * DIM * 3 * DIM, lqkvT + (size_t)l * 3 * DIM * DIM, DIM, 3 * DIM);
    tcvt(la_wo   + (size_t)l * DIM * DIM,     lwoT  + (size_t)l * DIM * DIM,     DIM, DIM);
    tcvt(lf_w1   + (size_t)l * DIM * FFD,     lw1T  + (size_t)l * FFD * DIM,     DIM, FFD);
    tcvt(lf_w2   + (size_t)l * FFD * DIM,     lw2T  + (size_t)l * DIM * FFD,     FFD, DIM);
  }
  tcvt(w_logit, wlT, DIM, VOC);

  // ---- embeddings ----
  embed_kernel<<<NTOK, TB, 0, stream>>>(x_ids, tok_emb, pos_emb, e_f32, DIM);
  cvt_bf16_kernel<<<(unsigned)(((size_t)NPRE * DIM + TB - 1) / TB), TB, 0, stream>>>(
      e_f32, pre_bf, (size_t)NPRE * DIM);

  const float scale = 0.125f; // DH^-0.5

  auto run_heads = [&](int Nk, int offset) {
    for (int h = 0; h < NH; ++h) {
      // S = Q_h K_h^T   (A: rows of Q slice; B: key rows, K-contiguous)
      gemm(q_bf + h * DHD, DIM, k_all + h * DHD, DIM, Sf, Nk,
           nullptr, nullptr, 0, NLAT, Nk, DHD, 0);
      softmax_bf16_kernel<<<NLAT, TB, 0, stream>>>(Sf, Pbf, Nk, offset, scale);
      // O_h = P * V_h  via NT with V^T rows (h*64 .. h*64+63)
      gemm(Pbf, Nk, vT + (size_t)h * DHD * Nk, Nk, attn_bf + h * DHD, DIM,
           nullptr, nullptr, 0, NLAT, DHD, Nk, F_BF16);
    }
  };

  // ================= cross-attention block =================
  ln_bf16_kernel<<<NLAT, TB, 0, stream>>>(hbuf, ca_ln_g, ca_ln_b, ln_bf, DIM);
  gemm(ln_bf, DIM, wqT, DIM, q_bf, DIM, nullptr, nullptr, 0, NLAT, DIM, DIM, F_BF16);
  // K/V of latents -> tail of concatenated key/value sequence
  gemm(ln_bf, DIM, wkvT, DIM, k_all + (size_t)NPRE * DIM, DIM,
       nullptr, nullptr, 0, NLAT, DIM, DIM, F_BF16);
  gemm(ln_bf, DIM, wkvT + (size_t)DIM * DIM, DIM, vT + NPRE, NTOK,
       nullptr, nullptr, 0, NLAT, DIM, DIM, F_BF16T);
  // K/V of prefix (raw embeddings, no LN) -> head of sequence
  gemm(pre_bf, DIM, wkvT, DIM, k_all, DIM, nullptr, nullptr, 0, NPRE, DIM, DIM, F_BF16);
  gemm(pre_bf, DIM, wkvT + (size_t)DIM * DIM, DIM, vT, NTOK,
       nullptr, nullptr, 0, NPRE, DIM, DIM, F_BF16T);
  run_heads(NTOK, NPRE);
  gemm(attn_bf, DIM, woT, DIM, hbuf, DIM, ca_bo, hbuf, DIM,
       NLAT, DIM, DIM, F_BIAS | F_RES);
  // FFN
  ln_bf16_kernel<<<NLAT, TB, 0, stream>>>(hbuf, cf_ln_g, cf_ln_b, ln_bf, DIM);
  gemm(ln_bf, DIM, w1T, DIM, ff1_bf, FFD, nullptr, nullptr, 0,
       NLAT, FFD, DIM, F_GELU | F_BF16);
  gemm(ff1_bf, FFD, w2T, FFD, hbuf, DIM, nullptr, hbuf, DIM,
       NLAT, DIM, FFD, F_RES);

  // ================= latent self-attention layers =================
  for (int l = 0; l < NDEPTH; ++l) {
    const bf16_t* qkvT = lqkvT + (size_t)l * 3 * DIM * DIM;
    ln_bf16_kernel<<<NLAT, TB, 0, stream>>>(hbuf, la_ln_g + l * DIM, la_ln_b + l * DIM, ln_bf, DIM);
    gemm(ln_bf, DIM, qkvT,                          DIM, q_bf,  DIM, nullptr, nullptr, 0, NLAT, DIM, DIM, F_BF16);
    gemm(ln_bf, DIM, qkvT + (size_t)DIM * DIM,      DIM, k_all, DIM, nullptr, nullptr, 0, NLAT, DIM, DIM, F_BF16);
    gemm(ln_bf, DIM, qkvT + (size_t)2 * DIM * DIM,  DIM, vT,   NLAT, nullptr, nullptr, 0, NLAT, DIM, DIM, F_BF16T);
    run_heads(NLAT, 0);
    gemm(attn_bf, DIM, lwoT + (size_t)l * DIM * DIM, DIM, hbuf, DIM,
         nullptr, hbuf, DIM, NLAT, DIM, DIM, F_RES);
    ln_bf16_kernel<<<NLAT, TB, 0, stream>>>(hbuf, lf_ln_g + l * DIM, lf_ln_b + l * DIM, ln_bf, DIM);
    gemm(ln_bf, DIM, lw1T + (size_t)l * FFD * DIM, DIM, ff1_bf, FFD,
         nullptr, nullptr, 0, NLAT, FFD, DIM, F_GELU | F_BF16);
    gemm(ff1_bf, FFD, lw2T + (size_t)l * DIM * FFD, FFD, hbuf, DIM,
         nullptr, hbuf, DIM, NLAT, DIM, FFD, F_RES);
  }

  // ================= logits =================
  cvt_bf16_kernel<<<(unsigned)(((size_t)NLAT * DIM + TB - 1) / TB), TB, 0, stream>>>(
      hbuf, h_bf, (size_t)NLAT * DIM);
  gemm(h_bf, DIM, wlT, DIM, (float*)d_out, VOC, nullptr, nullptr, 0,
       NLAT, VOC, DIM, 0);
}
